// MetaOptNet_48962627174542
// MI455X (gfx1250) — matrix operand order repaired and checked by reading, not compiled, not run on hardware
//
#include <hip/hip_runtime.h>

// ---------------------------------------------------------------------------
// MetaOptNet SVM-CS head for MI455X (gfx1250, wave32, WMMA)
//   B=128, n_way=5, n_shot=5 (ns=25), n_query=75, d=2560
//   Kernel1: bf16 WMMA fused Gram+compat GEMM (double-buffered LDS staging)
//   Kernel2: Mehrotra IPM; KKT solved via way-block-diagonal Schur complement
//            with LDS-resident Gauss-Jordan inverses (matvec-only solves)
// ---------------------------------------------------------------------------

typedef __attribute__((ext_vector_type(16))) __bf16 v16bf;
typedef __attribute__((ext_vector_type(8)))  float  v8f;

#define D_FEAT 2560
#define NS 25
#define NQ 75
#define NWAY 5
#define NDUAL 125
#define NEQ 25
#define KC 64                 // k per stage (2 WMMAs)
#define NSTAGE (D_FEAT / KC)  // 40
#define MAX_ITER 15
#define C_REG 0.1f
#define HSTRIDE 52            // row stride for 25x50 Gauss-Jordan workspaces

__device__ __forceinline__ unsigned short f32_to_bf16(float f) {
  unsigned int x = __float_as_uint(f);
  unsigned int lsb = (x >> 16) & 1u;
  x += 0x7FFFu + lsb;         // round to nearest even
  return (unsigned short)(x >> 16);
}

union FragBF16 { v16bf v; unsigned int u[8]; };

// ---------------------------------------------------------------------------
// Kernel 1: C (32x112) = S_pad (32 x d) * [S;Q]ᵀ, bf16 WMMA, f32 accumulate
// ---------------------------------------------------------------------------
__device__ __forceinline__ void stage_chunk(
    const float* __restrict__ Sb, const float* __restrict__ Qb,
    unsigned short* __restrict__ ldsA, unsigned short* __restrict__ ldsX,
    int k0, int tid) {
  for (int idx = tid; idx < 2304; idx += 512) {
    float4 v = {0.f, 0.f, 0.f, 0.f};
    unsigned short* dst;
    if (idx < 512) {
      int row = idx >> 4, q4 = idx & 15;
      if (row < NS) v = *(const float4*)(Sb + (size_t)row * D_FEAT + k0 + q4 * 4);
      dst = ldsA + row * KC + q4 * 4;
    } else {
      int j = idx - 512;
      int row = j >> 4, q4 = j & 15;
      if (row < NS)           v = *(const float4*)(Sb + (size_t)row * D_FEAT + k0 + q4 * 4);
      else if (row < NS + NQ) v = *(const float4*)(Qb + (size_t)(row - NS) * D_FEAT + k0 + q4 * 4);
      dst = ldsX + row * KC + q4 * 4;
    }
    uint2 packed;
    packed.x = ((unsigned)f32_to_bf16(v.y) << 16) | f32_to_bf16(v.x);
    packed.y = ((unsigned)f32_to_bf16(v.w) << 16) | f32_to_bf16(v.z);
    *(uint2*)dst = packed;
  }
}

__device__ __forceinline__ v8f compute_chunk(
    const unsigned short* __restrict__ ldsA, const unsigned short* __restrict__ ldsX,
    int mt, int nt, int lane, v8f acc) {
  const int hi = lane >> 4, l15 = lane & 15;
  const unsigned int* pa = (const unsigned int*)(ldsA + (mt * 16 + l15) * KC);
  const unsigned int* pb = (const unsigned int*)(ldsX + (nt * 16 + l15) * KC);
  #pragma unroll
  for (int c = 0; c < 2; ++c) {
    FragBF16 fa, fb;
    #pragma unroll
    for (int j = 0; j < 8; ++j)
      fa.u[j] = pa[c * 16 + ((j < 4) ? j : j + 4) + (hi ? 4 : 0)];
    #pragma unroll
    for (int j = 0; j < 8; ++j)
      fb.u[j] = pb[c * 16 + ((j < 4) ? j : j + 4) + (hi ? 4 : 0)];
    acc = __builtin_amdgcn_wmma_f32_16x16x32_bf16(
        false, fa.v, false, fb.v, (short)0, acc, false, false);
  }
  return acc;
}

__global__ __launch_bounds__(512)
void gram_compat_kernel(const float* __restrict__ S, const float* __restrict__ Q,
                        float* __restrict__ Kout, float* __restrict__ Cout) {
  const int b = blockIdx.x, tid = threadIdx.x;
  const int wv = tid >> 5, lane = tid & 31;

  __shared__ __align__(16) unsigned short ldsA[2][32 * KC];
  __shared__ __align__(16) unsigned short ldsX[2][112 * KC];

  const float* Sb = S + (size_t)b * NS * D_FEAT;
  const float* Qb = Q + (size_t)b * NQ * D_FEAT;

  const bool active = (wv < 14);
  const int mt = active ? (wv / 7) : 0;
  const int nt = active ? (wv % 7) : 0;

  v8f acc = {};

  stage_chunk(Sb, Qb, ldsA[0], ldsX[0], 0, tid);
  __syncthreads();

  for (int kc = 0; kc < NSTAGE; ++kc) {
    const int cur = kc & 1;
    if (kc + 2 < NSTAGE && tid < NS + NQ) {
      const float* p = (tid < NS) ? (Sb + (size_t)tid * D_FEAT)
                                  : (Qb + (size_t)(tid - NS) * D_FEAT);
      __builtin_prefetch(p + (kc + 2) * KC, 0, 1);
      __builtin_prefetch(p + (kc + 2) * KC + 32, 0, 1);
    }
    if (kc + 1 < NSTAGE)
      stage_chunk(Sb, Qb, ldsA[cur ^ 1], ldsX[cur ^ 1], (kc + 1) * KC, tid);
    if (active)
      acc = compute_chunk(ldsA[cur], ldsX[cur], mt, nt, lane, acc);
    __syncthreads();
  }

  if (active) {
    const int hi = lane >> 4, l15 = lane & 15;
    const int col = nt * 16 + l15;
    #pragma unroll
    for (int r = 0; r < 8; ++r) {
      int row = mt * 16 + (hi ? (r + 8) : r);
      if (row < NS) {
        float v = acc[r];
        if (col < NS)
          Kout[(size_t)b * NS * NS + row * NS + col] = v;
        else if (col < NS + NQ)
          Cout[(size_t)b * NS * NQ + row * NQ + (col - NS)] = v;
      }
    }
  }
}

// ---------------------------------------------------------------------------
// wave32 block reductions (256 threads = 8 waves)
// ---------------------------------------------------------------------------
__device__ __forceinline__ float wave_block_sum(float v, float* red, int tid) {
  #pragma unroll
  for (int off = 16; off > 0; off >>= 1) v += __shfl_down(v, off, 32);
  if ((tid & 31) == 0) red[tid >> 5] = v;
  __syncthreads();
  float r = (tid < 8) ? red[tid] : 0.0f;
  #pragma unroll
  for (int off = 4; off > 0; off >>= 1) r += __shfl_down(r, off, 32);
  if (tid == 0) red[0] = r;
  __syncthreads();
  r = red[0];
  __syncthreads();
  return r;
}
__device__ __forceinline__ float wave_block_min(float v, float* red, int tid) {
  #pragma unroll
  for (int off = 16; off > 0; off >>= 1) v = fminf(v, __shfl_down(v, off, 32));
  if ((tid & 31) == 0) red[tid >> 5] = v;
  __syncthreads();
  float r = (tid < 8) ? red[tid] : 3.0e38f;
  #pragma unroll
  for (int off = 4; off > 0; off >>= 1) r = fminf(r, __shfl_down(r, off, 32));
  if (tid == 0) red[0] = r;
  __syncthreads();
  r = red[0];
  __syncthreads();
  return r;
}

// KKT solve [[H,A'],[A,0]][dz;dnu]=[r1;-rp] via Schur; matvec-only (inverses).
__device__ void kkt_solve(const float* __restrict__ HW, const float* __restrict__ SW,
                          const float* __restrict__ r1, const float* __restrict__ rp,
                          float* yv, float* t25, float* dnu, float* dzo, int tid) {
  if (tid < NDUAL) {                       // y = H^{-1} r1 (block-diag over w)
    int si = tid / NWAY, w = tid % NWAY;
    const float* Hi = HW + (w * 25 + si) * HSTRIDE + 25;
    float a = 0.f;
    for (int sp = 0; sp < 25; ++sp) a += Hi[sp] * r1[sp * NWAY + w];
    yv[tid] = a;
  }
  __syncthreads();
  if (tid < NEQ) {                         // A y + r_p
    float a = rp[tid];
    for (int w = 0; w < NWAY; ++w) a += yv[tid * NWAY + w];
    t25[tid] = a;
  }
  __syncthreads();
  if (tid < NEQ) {                         // dnu = S^{-1} (A y + r_p)
    const float* Si = SW + tid * HSTRIDE + 25;
    float a = 0.f;
    for (int q = 0; q < 25; ++q) a += Si[q] * t25[q];
    dnu[tid] = a;
  }
  __syncthreads();
  if (tid < NDUAL) {                       // dz = y - H^{-1} A' dnu
    int si = tid / NWAY, w = tid % NWAY;
    const float* Hi = HW + (w * 25 + si) * HSTRIDE + 25;
    float a = 0.f;
    for (int sp = 0; sp < 25; ++sp) a += Hi[sp] * dnu[sp];
    dzo[tid] = yv[tid] - a;
  }
  __syncthreads();
}

// ---------------------------------------------------------------------------
// Kernel 2: Mehrotra predictor-corrector IPM, one block / batch, all LDS.
// ---------------------------------------------------------------------------
__global__ __launch_bounds__(256)
void qp_kernel(const float* __restrict__ Kws, const int* __restrict__ labels,
               float* __restrict__ zOut) {
  const int b = blockIdx.x, tid = threadIdx.x;

  __shared__ float Ks[NS * NS];
  __shared__ float HW[5 * 25 * HSTRIDE];
  __shared__ float SW[25 * HSTRIDE];
  __shared__ float oh[NDUAL], z[NDUAL], s[NDUAL], lam[NDUAL], nu[NEQ];
  __shared__ float Dls[NDUAL], rd[NDUAL], rp[NEQ], rhs1[NDUAL];
  __shared__ float yv[NDUAL], dza[NDUAL], dla[NDUAL], rc[NDUAL], dzc[NDUAL];
  __shared__ float t25[NEQ], dnu[NEQ];
  __shared__ float red[8];

  for (int i = tid; i < NS * NS; i += 256) Ks[i] = Kws[(size_t)b * NS * NS + i];
  for (int i = tid; i < NDUAL; i += 256) {
    int si = i / NWAY, wi = i % NWAY;
    float o = (labels[b * NS + si] == wi) ? 1.0f : 0.0f;
    oh[i] = o;
    z[i] = C_REG * o - 1.0f;
    s[i] = 1.0f;
    lam[i] = 1.0f;
  }
  for (int i = tid; i < NEQ; i += 256) nu[i] = 0.0f;
  __syncthreads();

  for (int iter = 0; iter < MAX_ITER; ++iter) {
    if (tid < NDUAL) {
      int si = tid / NWAY, wi = tid % NWAY;
      Dls[tid] = lam[tid] / s[tid];
      float a = z[tid];
      for (int sj = 0; sj < NS; ++sj) a += Ks[si * NS + sj] * z[sj * NWAY + wi];
      rd[tid] = a - oh[tid] + lam[tid] + nu[si];
    }
    if (tid < NEQ) {
      float a = 0.f;
      for (int wj = 0; wj < NWAY; ++wj) a += z[tid * NWAY + wj];
      rp[tid] = a;
    }
    float mu = wave_block_sum((tid < NDUAL) ? s[tid] * lam[tid] : 0.f, red, tid)
               * (1.0f / (float)NDUAL);

    // build [H_w | I] and Gauss-Jordan (1 barrier/step; SPD -> no pivoting)
    for (int idx = tid; idx < 5 * 25 * 50; idx += 256) {
      int w = idx / 1250, r = idx % 1250, i = r / 50, j = r % 50;
      float v;
      if (j < 25) {
        v = Ks[i * NS + j];
        if (i == j) v += 1.0f + Dls[i * NWAY + w];
      } else v = (j - 25 == i) ? 1.0f : 0.0f;
      HW[(w * 25 + i) * HSTRIDE + j] = v;
    }
    __syncthreads();
    for (int k = 0; k < 25; ++k) {
      if (tid < 125) {
        int w = tid / 25, i = tid % 25;
        if (i != k) {
          float* Mi = HW + (w * 25 + i) * HSTRIDE;
          const float* Mk = HW + (w * 25 + k) * HSTRIDE;
          float f = Mi[k] / Mk[k];
          for (int j = k; j < 25; ++j)       Mi[j] -= f * Mk[j];
          for (int j = 25; j <= 25 + k; ++j) Mi[j] -= f * Mk[j];
        }
      }
      __syncthreads();
    }
    for (int idx = tid; idx < 5 * 25 * 25; idx += 256) {
      int w = idx / 625, r = idx % 625, i = r / 25, j = r % 25;
      float* Mi = HW + (w * 25 + i) * HSTRIDE;
      Mi[25 + j] /= Mi[i];
    }
    __syncthreads();

    // Schur S = sum_w H_w^{-1}; invert
    for (int idx = tid; idx < 25 * 50; idx += 256) {
      int i = idx / 50, j = idx % 50;
      float v;
      if (j < 25) {
        v = 0.f;
        for (int w = 0; w < 5; ++w) v += HW[(w * 25 + i) * HSTRIDE + 25 + j];
      } else v = (j - 25 == i) ? 1.0f : 0.0f;
      SW[i * HSTRIDE + j] = v;
    }
    __syncthreads();
    for (int k = 0; k < 25; ++k) {
      if (tid < 25 && tid != k) {
        float* Mi = SW + tid * HSTRIDE;
        const float* Mk = SW + k * HSTRIDE;
        float f = Mi[k] / Mk[k];
        for (int j = k; j < 25; ++j)       Mi[j] -= f * Mk[j];
        for (int j = 25; j <= 25 + k; ++j) Mi[j] -= f * Mk[j];
      }
      __syncthreads();
    }
    for (int idx = tid; idx < 625; idx += 256) {
      int i = idx / 25, j = idx % 25;
      SW[i * HSTRIDE + 25 + j] /= SW[i * HSTRIDE + i];
    }
    __syncthreads();

    // predictor
    if (tid < NDUAL) rhs1[tid] = lam[tid] - rd[tid];
    __syncthreads();
    kkt_solve(HW, SW, rhs1, rp, yv, t25, dnu, dza, tid);
    if (tid < NDUAL) dla[tid] = -lam[tid] + Dls[tid] * dza[tid];
    float bmin = 1.0f;
    if (tid < NDUAL) {
      float dsv = -dza[tid];
      if (dsv < 0.f) bmin = fminf(bmin, -s[tid] / dsv);
      if (dla[tid] < 0.f) bmin = fminf(bmin, -lam[tid] / dla[tid]);
    }
    float a_aff = wave_block_min(bmin, red, tid);
    float part = 0.f;
    if (tid < NDUAL)
      part = (s[tid] - a_aff * dza[tid]) * (lam[tid] + a_aff * dla[tid]);
    float mu_aff = wave_block_sum(part, red, tid) * (1.0f / (float)NDUAL);
    float ratio = mu_aff / mu;
    float sigma = ratio * ratio * ratio;

    // corrector
    if (tid < NDUAL) {
      float v = sigma * mu + dza[tid] * dla[tid];
      rc[tid] = v;
      rhs1[tid] = lam[tid] - rd[tid] - v / s[tid];
    }
    __syncthreads();
    kkt_solve(HW, SW, rhs1, rp, yv, t25, dnu, dzc, tid);

    float bmin2 = 1.0f;
    if (tid < NDUAL) {
      float dzv = dzc[tid];
      float dlv = rc[tid] / s[tid] - lam[tid] + Dls[tid] * dzv;
      dla[tid] = dlv;
      if (-dzv < 0.f) bmin2 = fminf(bmin2, -s[tid] / (-dzv));
      if (dlv < 0.f)  bmin2 = fminf(bmin2, -lam[tid] / dlv);
    }
    float alpha = 0.995f * wave_block_min(bmin2, red, tid);

    if (tid < NDUAL) {
      float dzv = dzc[tid];
      z[tid]   += alpha * dzv;
      s[tid]   -= alpha * dzv;
      lam[tid] += alpha * dla[tid];
    }
    if (tid < NEQ) nu[tid] += alpha * dnu[tid];
    __syncthreads();
  }

  for (int i = tid; i < NDUAL; i += 256)
    zOut[(size_t)b * NDUAL + i] = z[i];
}

// ---------------------------------------------------------------------------
// Kernel 3: logits[b][q][w] = scale * sum_s z[b][s*5+w] * compat[b][s][q]
// ---------------------------------------------------------------------------
__global__ __launch_bounds__(256)
void logits_kernel(const float* __restrict__ zWs, const float* __restrict__ Cws,
                   const float* __restrict__ scale, float* __restrict__ out) {
  const int b = blockIdx.x, tid = threadIdx.x;
  __shared__ float zsh[NDUAL];
  for (int i = tid; i < NDUAL; i += 256) zsh[i] = zWs[(size_t)b * NDUAL + i];
  __syncthreads();
  const float sc = scale[0];
  const float* Cb = Cws + (size_t)b * NS * NQ;
  for (int idx = tid; idx < NQ * NWAY; idx += 256) {
    int q = idx / NWAY, w = idx % NWAY;
    float a = 0.f;
    for (int si = 0; si < NS; ++si)
      a += zsh[si * NWAY + w] * Cb[si * NQ + q];
    out[(size_t)b * NQ * NWAY + idx] = a * sc;
  }
}

// ---------------------------------------------------------------------------
extern "C" void kernel_launch(void* const* d_in, const int* in_sizes, int n_in,
                              void* d_out, int out_size, void* d_ws, size_t ws_size,
                              hipStream_t stream) {
  const float* query   = (const float*)d_in[0];   // (B, 75, 2560) f32
  const float* support = (const float*)d_in[1];   // (B, 25, 2560) f32
  const float* scale   = (const float*)d_in[2];   // (1,) f32
  const int*   labels  = (const int*)d_in[3];     // (B, 25) i32
  (void)n_in; (void)out_size; (void)ws_size;

  const int B = in_sizes[1] / (NS * D_FEAT);      // 128

  float* ws  = (float*)d_ws;
  float* Kws = ws;                                // B*625
  float* Cws = Kws + (size_t)B * NS * NS;         // B*1875
  float* zWs = Cws + (size_t)B * NS * NQ;         // B*125

  gram_compat_kernel<<<B, 512, 0, stream>>>(support, query, Kws, Cws);
  qp_kernel<<<B, 256, 0, stream>>>(Kws, labels, zWs);
  logits_kernel<<<B, 256, 0, stream>>>(zWs, Cws, scale, (float*)d_out);
}
